// CDB_34333968564293
// MI455X (gfx1250) — matrix operand order repaired and checked by reading, not compiled, not run on hardware
//
#include <hip/hip_runtime.h>
#include <hip/hip_bf16.h>

typedef __bf16 bf16;
typedef bf16 v16bf __attribute__((ext_vector_type(16)));
typedef bf16 v16bf_a16 __attribute__((ext_vector_type(16), aligned(16)));
typedef float v8f  __attribute__((ext_vector_type(8)));
typedef float v8f_a16 __attribute__((ext_vector_type(8), aligned(16)));
typedef float v2f  __attribute__((ext_vector_type(2)));

#define BATCH 4
#define CH    64
#define HH    256
#define WW    256

// ---------------------------------------------------------------- helpers
// Single contiguous 32B fragment load (backend: 2x ds_load_b128, adjacent regs)
__device__ __forceinline__ v16bf ld_frag(const bf16* p) {
  return *(const v16bf_a16*)p;
}
// A-side channel permute: store 32-ch chunk as [0-7, 16-23, 8-15, 24-31] so that
// lane-half h's hardware-K set {8h..8h+7, 16+8h..23+8h} is stored [16h,16h+16).
// dword index within 32: swap bits 2<->3
__device__ __forceinline__ int permdw(int cp) {
  return (cp & ~12) | ((cp & 4) << 1) | ((cp & 8) >> 1);
}
// same permute at 8-channel (16B) granularity: swap low two group bits
__device__ __forceinline__ int permg8(int c8) {
  int g = c8 & 3;
  return (c8 & 4) | ((g & 1) << 1) | (g >> 1);
}
// CDNA5 async global->LDS copy (16B), GV mode; tracked on ASYNCcnt
__device__ __forceinline__ void async_ld16(void* lds, const void* g) {
  unsigned ldsoff = (unsigned)(unsigned long long)lds;   // low 32 bits = LDS offset
  asm volatile("global_load_async_to_lds_b128 %0, %1, off"
               :: "v"(ldsoff), "v"((unsigned long long)g) : "memory");
}
__device__ __forceinline__ void wait_async() {
  asm volatile("s_wait_asynccnt 0x0" ::: "memory");
}

// ---------------------------------------------------------------- P0: weights
// wt1/wt2 : [64][576] bf16, k = tap*64+ci   (tap = ky*3+kx)
// wt3     : [9][64][64] bf16 : (j, c, ci) = w3[c*9+j][ci]
__global__ void prep_weights(const float* __restrict__ w1, const float* __restrict__ w2,
                             const float* __restrict__ w3,
                             bf16* __restrict__ wt1, bf16* __restrict__ wt2,
                             bf16* __restrict__ wt3) {
  int idx = blockIdx.x * 256 + threadIdx.x;
  if (idx >= 3 * 36864) return;
  int sec = idx / 36864, r = idx % 36864;
  if (sec < 2) {
    const float* w = sec ? w2 : w1;
    bf16* o = sec ? wt2 : wt1;
    int n = r / 576, k = r % 576;
    int tap = k / 64, ci = k % 64;
    o[n * 576 + k] = (bf16)w[(size_t)(n * 64 + ci) * 9 + tap];
  } else {
    int j = r / 4096, rem = r % 4096;
    int c = rem / 64, ci = rem % 64;
    wt3[(j * 64 + c) * 64 + ci] = (bf16)w3[(size_t)(c * 9 + j) * 64 + ci];
  }
}

// ---------------------------------------------------------------- P1: inputs
// s = bf16(x+y) NHWC
__global__ void prep_inputs(const float* __restrict__ x, const float* __restrict__ y,
                            bf16* __restrict__ s) {
  __shared__ float ls[64 * 32];
  const int t = threadIdx.x;
  const int w0 = blockIdx.x * 32, h = blockIdx.y, bb = blockIdx.z;
  for (int i = t; i < 2048; i += 256) {
    int c = i >> 5, w = i & 31;
    size_t gi = ((size_t)(bb * 64 + c) * 256 + h) * 256 + w0 + w;
    ls[i] = x[gi] + y[gi];
  }
  __syncthreads();
  for (int i = t; i < 2048; i += 256) {
    int w = i >> 6, c = i & 63;
    s[((size_t)(bb * 256 + h) * 256 + w0 + w) * 64 + c] = (bf16)ls[c * 32 + w];
  }
}

// ---------------------------------------------------------------- conv3x3 (implicit GEMM)
// Tile: 16x16 positions x 64 ch, 8 waves, 2 M-tiles per wave.
template <bool LEAKY>
__global__ __launch_bounds__(256) void conv3x3_wmma(const bf16* __restrict__ in,
                                                    const bf16* __restrict__ wt,
                                                    bf16* __restrict__ out) {
  __shared__ __align__(32) bf16 sh_in[18 * 18 * 64];   // halo tile, channels in natural order
  __shared__ __align__(32) bf16 sh_w[64 * 576];        // B^T: [n][k]
  __shared__ __align__(16) bf16 sh_out[256 * 64];

  const int t = threadIdx.x;
  const int tile = blockIdx.x;
  const int tx = tile & 15, ty = (tile >> 4) & 15, bb = tile >> 8;
  const int h0 = ty * 16, w0 = tx * 16;

  // weights -> LDS via CDNA5 async copy (4608 x 16B)
  for (int i = t; i < 4608; i += 256)
    async_ld16(sh_w + i * 8, (const void*)(wt + i * 8));
  // input halo tile -> LDS (natural channel order), zero-padded borders (10368 dwords)
  for (int i = t; i < 18 * 18 * 32; i += 256) {
    int cp = i & 31;
    int xx = (i >> 5) % 18;
    int yy = (i >> 5) / 18;
    int gh = h0 - 1 + yy, gw = w0 - 1 + xx;
    unsigned v = 0u;
    if ((unsigned)gh < 256u && (unsigned)gw < 256u)
      v = *(const unsigned*)(in + ((((size_t)bb * 256 + gh) * 256 + gw) * 64 + cp * 2));
    *(unsigned*)(sh_in + ((yy * 18 + xx) * 64 + cp * 2)) = v;
  }
  wait_async();
  __syncthreads();

  const int wv = t >> 5, lane = t & 31;
  const int rm = lane & 15, hf = lane >> 4;

  v8f acc[2][4] = {};
#pragma unroll 1
  for (int kc = 0; kc < 18; ++kc) {           // K = 576 = 18 x 32
    const int tap = kc >> 1, ci0 = (kc & 1) << 5;
    const int ky = tap / 3, kx = tap % 3;
    v16bf a[2];
#pragma unroll
    for (int mt = 0; mt < 2; ++mt) {
      const int ay = wv * 2 + mt + ky;
      a[mt] = ld_frag(sh_in + ((ay * 18 + (rm + kx)) * 64 + ci0 + 16 * hf));
    }
#pragma unroll
    for (int nt = 0; nt < 4; ++nt) {
      v16bf b = ld_frag(sh_w + (nt * 16 + rm) * 576 + kc * 32 + 16 * hf);
#pragma unroll
      for (int mt = 0; mt < 2; ++mt)
        acc[mt][nt] = __builtin_amdgcn_wmma_f32_16x16x32_bf16(false, a[mt], false, b,
                                                              (short)0, acc[mt][nt], false, false);
    }
  }

#pragma unroll
  for (int mt = 0; mt < 2; ++mt)
#pragma unroll
    for (int nt = 0; nt < 4; ++nt)
#pragma unroll
      for (int e = 0; e < 8; ++e) {
        float f = acc[mt][nt][e];
        if (LEAKY) f = f > 0.f ? f : 0.01f * f;
        int pos = (wv * 2 + mt) * 16 + (e + 8 * hf);
        sh_out[pos * 64 + nt * 16 + rm] = (bf16)f;
      }
  __syncthreads();

  const uint4* src = (const uint4*)sh_out;
  for (int i = t; i < 2048; i += 256) {       // coalesced NHWC store
    int c8 = i & 7, p = i >> 3;
    int py = p >> 4, px = p & 15;
    *(uint4*)(out + ((((size_t)bb * 256 + h0 + py) * 256 + (w0 + px)) * 64 + c8 * 8)) = src[i];
  }
}

// ---------------------------------------------------------------- K3: conv1x1 + dynamic filter + residual
__global__ __launch_bounds__(256) void fused_filter_wmma(const bf16* __restrict__ t2,
                                                         const float* __restrict__ xf,
                                                         const bf16* __restrict__ wt3,
                                                         float* __restrict__ outp) {
  __shared__ __align__(32) bf16  sh_a[256 * 64];       // t2 tile, natural channel order (32768 B)
  __shared__ __align__(16) float sh_xf[64 * 18 * 20];  // x patches f32 [c][y][20] (92160 B)
  __shared__ __align__(32) bf16  sh_w[9 * 64 * 64];    // per-tap B^T (73728 B)
  __shared__ __align__(16) float sh_o[64 * 256];       // x f32 in / result out (65536 B)

  const int t = threadIdx.x;
  const int tile = blockIdx.x;
  const int tx = tile & 15, ty = (tile >> 4) & 15, bb = tile >> 8;
  const int h0 = ty * 16, w0 = tx * 16;

  // t2 tile (natural channel order) + wt3 -> LDS via async copies
  for (int i = t; i < 2048; i += 256) {
    int c8 = i & 7, p = i >> 3, py = p >> 4, px = p & 15;
    async_ld16(sh_a + p * 64 + c8 * 8,
               (const void*)(t2 + ((((size_t)bb * 256 + h0 + py) * 256 + w0 + px) * 64 + c8 * 8)));
  }
  for (int i = t; i < 4608; i += 256)
    async_ld16(sh_w + i * 8, (const void*)(wt3 + i * 8));
  // x patches f32 (zero-padded halo): [c][y 0..17][x 0..17] (cols 18,19 unused)
  for (int i = t; i < 64 * 18 * 18; i += 256) {
    int xx = i % 18, yy = (i / 18) % 18, c = i / 324;
    int gh = h0 - 1 + yy, gw = w0 - 1 + xx;
    float v = 0.f;
    if ((unsigned)gh < 256u && (unsigned)gw < 256u)
      v = xf[(((size_t)bb * 64 + c) * 256 + gh) * 256 + gw];
    sh_xf[(c * 18 + yy) * 20 + xx] = v;
  }
  // f32 x tile (residual) -> LDS, NCHW coalesced
  for (int i = t; i < 16384; i += 256) {
    int px = i & 15, py = (i >> 4) & 15, c = i >> 8;
    sh_o[c * 256 + py * 16 + px] =
        xf[(((size_t)bb * 64 + c) * 256 + h0 + py) * 256 + w0 + px];
  }
  wait_async();
  __syncthreads();

  const int wv = t >> 5, lane = t & 31;
  const int rm = lane & 15, hf = lane >> 4;

  v16bf a0[2], a1[2];                         // K = 64 ci = 2 chunks, 2 M-tiles
#pragma unroll
  for (int mt = 0; mt < 2; ++mt) {
    const bf16* ap = sh_a + ((wv * 2 + mt) * 16 + rm) * 64;
    a0[mt] = ld_frag(ap + 16 * hf);
    a1[mt] = ld_frag(ap + 32 + 16 * hf);
  }

  v8f oacc[2][4] = {};
#pragma unroll 1
  for (int jy = 0; jy < 3; ++jy) {            // rolled: bounds the scheduler's hoist window
#pragma unroll
    for (int nt = 0; nt < 4; ++nt) {
      const int c = nt * 16 + rm;
      v8f xa[2]; v2f xb[2];                   // x row f32 slice [8h..8h+10), shared over jx
#pragma unroll
      for (int mt = 0; mt < 2; ++mt) {
        const float* rp = sh_xf + (c * 18 + (wv * 2 + mt + jy)) * 20 + 8 * hf;
        xa[mt] = *(const v8f_a16*)rp;
        xb[mt] = *(const v2f*)(rp + 8);
      }
#pragma unroll
      for (int jx = 0; jx < 3; ++jx) {
        const bf16* bp = sh_w + ((jy * 3 + jx) * 64 + c) * 64;
        v16bf b0 = ld_frag(bp + 16 * hf);
        v16bf b1 = ld_frag(bp + 32 + 16 * hf);
#pragma unroll
        for (int mt = 0; mt < 2; ++mt) {
          v8f d = {};
          d = __builtin_amdgcn_wmma_f32_16x16x32_bf16(false, a0[mt], false, b0, (short)0, d, false, false);
          d = __builtin_amdgcn_wmma_f32_16x16x32_bf16(false, a1[mt], false, b1, (short)0, d, false, false);
#pragma unroll
          for (int e = 0; e < 8; ++e) {       // f32 extracts are register refs -> plain v_fmac_f32
            const int k = e + jx;
            float xv = (k < 8) ? xa[mt][k] : xb[mt][k - 8];
            oacc[mt][nt][e] += xv * d[e];
          }
        }
      }
    }
  }

#pragma unroll
  for (int mt = 0; mt < 2; ++mt)
#pragma unroll
    for (int nt = 0; nt < 4; ++nt) {
      int c = nt * 16 + rm;
#pragma unroll
      for (int e = 0; e < 8; ++e) {
        int pos = (wv * 2 + mt) * 16 + (e + 8 * hf);
        sh_o[c * 256 + pos] += oacc[mt][nt][e];   // + x residual (f32), unique lane ownership
      }
    }
  __syncthreads();
  for (int i = t; i < 16384; i += 256) {      // coalesced NCHW f32 store
    int px = i & 15, py = (i >> 4) & 15, c = i >> 8;
    outp[(((size_t)bb * 64 + c) * 256 + h0 + py) * 256 + w0 + px] =
        sh_o[c * 256 + py * 16 + px];
  }
}

// ---------------------------------------------------------------- launch
extern "C" void kernel_launch(void* const* d_in, const int* in_sizes, int n_in,
                              void* d_out, int out_size, void* d_ws, size_t ws_size,
                              hipStream_t stream) {
  const float* x  = (const float*)d_in[0];
  const float* y  = (const float*)d_in[1];
  const float* w1 = (const float*)d_in[2];
  const float* w2 = (const float*)d_in[3];
  const float* w3 = (const float*)d_in[4];
  float* out = (float*)d_out;

  const size_t NPIX = (size_t)BATCH * HH * WW;          // 262144
  const size_t BF_TENSOR = NPIX * CH * sizeof(bf16);    // 33,554,432 B

  char* ws = (char*)d_ws;
  bf16* s_b  = (bf16*)(ws);
  bf16* t1_b = (bf16*)(ws + BF_TENSOR);
  bf16* t2_b = (bf16*)(ws + 2 * BF_TENSOR);
  bf16* wt1  = (bf16*)(ws + 3 * BF_TENSOR);
  bf16* wt2  = wt1 + 64 * 576;
  bf16* wt3  = wt2 + 64 * 576;

  prep_weights<<<(3 * 36864 + 255) / 256, 256, 0, stream>>>(w1, w2, w3, wt1, wt2, wt3);
  prep_inputs<<<dim3(WW / 32, HH, BATCH), 256, 0, stream>>>(x, y, s_b);

  const int ntiles = BATCH * (HH / 16) * (WW / 16);     // 1024
  conv3x3_wmma<true ><<<ntiles, 256, 0, stream>>>(s_b,  wt1, t1_b);
  conv3x3_wmma<false><<<ntiles, 256, 0, stream>>>(t1_b, wt2, t2_b);
  fused_filter_wmma<<<ntiles, 256, 0, stream>>>(t2_b, x, wt3, out);
}
